// ElectronicEmbedding_61022895342002
// MI455X (gfx1250) — matrix operand order repaired and checked by reading, not compiled, not run on hardware
//
#include <hip/hip_runtime.h>
#include <hip/hip_bf16.h>

typedef __attribute__((ext_vector_type(16))) _Float16 v16h;
typedef __attribute__((ext_vector_type(8)))  _Float16 v8h;
typedef __attribute__((ext_vector_type(8)))  float    v8f;
typedef __attribute__((ext_vector_type(4)))  float    v4f;

#define NROWS   1000000
#define FDIM    128
#define NBATCH  4096
#define TILE    64
#define NTILES  (NROWS / TILE)      /* 15625 exactly */
#define USTRIDE 136                  /* halves: 128 + 8 pad -> bank-conflict-free */
#define RSQRT_F 0.08838834764831845f
#define EPS     1e-8f

// silu with fast hardware reciprocal (v_rcp_f32, ~1 ulp) instead of the
// IEEE div_scale/div_fmas/div_fixup sequence.
__device__ __forceinline__ float siluf(float x) {
  return x * __builtin_amdgcn_rcpf(1.0f + __expf(-x));
}
__device__ __forceinline__ float softplusf(float x) {
  // stable log(1+e^x)
  return fmaxf(x, 0.0f) + log1pf(__expf(-fabsf(x)));
}

// ---------------------------------------------------------------------------
// prep0: t0[j] = sum_f Wq[f,j]*Wk[f,0]; t1[j] = sum_f Wq[f,j]*Wk[f,1]
//        u01[c] = sum_f Wk[f,c]*bq[f]
__global__ void k_prep0(const float* __restrict__ Wq, const float* __restrict__ bq,
                        const float* __restrict__ Wk,
                        float* __restrict__ t0a, float* __restrict__ t1a,
                        float* __restrict__ u01) {
  int j = threadIdx.x;                       // 0..127
  float t0 = 0.f, t1 = 0.f;
  for (int f = 0; f < FDIM; ++f) {
    float wq = Wq[f * FDIM + j];
    t0 = fmaf(wq, Wk[f * 2 + 0], t0);
    t1 = fmaf(wq, Wk[f * 2 + 1], t1);
  }
  t0a[j] = t0; t1a[j] = t1;
  if (j < 2) {
    float u = 0.f;
    for (int f = 0; f < FDIM; ++f) u = fmaf(Wk[f * 2 + j], bq[f], u);
    u01[j] = u;
  }
}

// ---------------------------------------------------------------------------
// prep1: per-batch crow/vrow/dvec; zero anorm.
__global__ void k_prep1(const float* __restrict__ E, const float* __restrict__ Wv,
                        const float* __restrict__ t0a, const float* __restrict__ t1a,
                        const float* __restrict__ u01,
                        float* __restrict__ crow, float* __restrict__ vrow,
                        float* __restrict__ dvec, float* __restrict__ anorm) {
  int idx = blockIdx.x * blockDim.x + threadIdx.x;  // B*F threads
  int b = idx >> 7, j = idx & 127;
  float Eb = E[b];
  float e0 = fmaxf(Eb, 0.f), e1 = fmaxf(-Eb, 0.f);
  float s0 = e0 / fmaxf(e0, 1.f), s1 = e1 / fmaxf(e1, 1.f);
  crow[idx] = s0 * t0a[j] + s1 * t1a[j];
  vrow[idx] = e0 * Wv[j * 2 + 0] + e1 * Wv[j * 2 + 1];
  if (j == 0) {
    dvec[b]  = s0 * u01[0] + s1 * u01[1];
    anorm[b] = 0.f;
  }
}

// ---------------------------------------------------------------------------
// pack W1/W2 into f16 WMMA B-fragment layout (B is 32x16 per frag, lane%16 = N,
// v16h elems = K in table order):  pack[((w*4+kc)*32 + l)*16 + e] =
//   W[(w*16+(l&15))*128 + kc*32 + 8*(l>>4) + (e&7) + ((e&8)<<1)]
__global__ void k_pack(const float* __restrict__ W1, const float* __restrict__ W2,
                       _Float16* __restrict__ W1p, _Float16* __restrict__ W2p) {
  int idx = blockIdx.x * blockDim.x + threadIdx.x;  // 0..16383
  const float* W   = (idx < 8192) ? W1  : W2;
  _Float16*    dst = (idx < 8192) ? W1p : W2p;
  int rem = idx & 8191;
  int e  = rem & 15;
  int l  = (rem >> 4) & 31;
  int kc = (rem >> 9) & 3;
  int w  = (rem >> 11) & 7;
  dst[rem] = (_Float16)W[(w * 16 + (l & 15)) * FDIM + kc * 32 + ((l >> 4) << 3) + (e & 7) + ((e & 8) << 1)];
}

// ---------------------------------------------------------------------------
// pass 1: dot[i] = (x[i].c[b] + d[b])/sqrt(F); a = softplus; anorm = segsum(a)
// One wave handles 128 contiguous rows (sorted batch_seg -> few atomics).
// x is a 512MB read-once stream -> non-temporal so it doesn't evict the
// L2-resident crow/vrow/weight tables.
__global__ void __launch_bounds__(256)
k_dot(const float* __restrict__ x, const int* __restrict__ bseg,
      const float* __restrict__ crow, const float* __restrict__ dvec,
      float* __restrict__ a_arr, float* __restrict__ anorm) {
  const int lane = threadIdx.x & 31;
  const int wv = blockIdx.x * 8 + (threadIdx.x >> 5);
  long r0 = (long)wv * 128;
  long r1 = r0 + 128; if (r1 > NROWS) r1 = NROWS;
  float accum = 0.f; int bprev = -1;
  for (long i = r0; i < r1; ++i) {
    int b = bseg[i];
    v4f xv = __builtin_nontemporal_load((const v4f*)x + i * 32 + lane);
    v4f cv = *((const v4f*)crow + (long)b * 32 + lane);   // L2-hot (2MB)
    float p = xv[0] * cv[0] + xv[1] * cv[1] + xv[2] * cv[2] + xv[3] * cv[3];
#pragma unroll
    for (int off = 16; off >= 1; off >>= 1) p += __shfl_xor(p, off, 32);
    float a = softplusf((p + dvec[b]) * RSQRT_F);
    if (lane == 0) {
      a_arr[i] = a;
      if (b != bprev) {
        if (bprev >= 0) atomicAdd(anorm + bprev, accum);
        bprev = b; accum = 0.f;
      }
      accum += a;
    }
  }
  if (lane == 0 && bprev >= 0) atomicAdd(anorm + bprev, accum);
}

// ---------------------------------------------------------------------------
// pass 2: h = attn*v ; out = h + W2(silu(W1(silu(h))))   [WMMA f16, f32 acc]
// 64-row tiles; wave w owns output columns [16w,16w+16); B-frags persistent.
__global__ void __launch_bounds__(256)
k_mlp(const float* __restrict__ a_arr, const float* __restrict__ anorm,
      const int* __restrict__ bseg, const float* __restrict__ vrow,
      const _Float16* __restrict__ W1p, const _Float16* __restrict__ W2p,
      float* __restrict__ out) {
  __shared__ __align__(16) _Float16 u_t[TILE * USTRIDE];  // silu(h), f16
  __shared__ __align__(16) _Float16 z_t[TILE * USTRIDE];  // silu(gemm1), f16
  __shared__ float attn_s[TILE];
  __shared__ int   bseg_s[TILE];

  const int tid  = threadIdx.x;
  const int lane = tid & 31;
  const int w    = tid >> 5;        // wave id 0..7
  const int n0   = w * 16;          // this wave's output-column block

  // Persistent B-fragments: 4 K-chunks per matrix, 32B contiguous per lane.
  v16h bw1[4], bw2[4];
#pragma unroll
  for (int kc = 0; kc < 4; ++kc) {
    bw1[kc] = *(const v16h*)(W1p + (size_t)((w * 4 + kc) * 32 + lane) * 16);
    bw2[kc] = *(const v16h*)(W2p + (size_t)((w * 4 + kc) * 32 + lane) * 16);
  }

  for (int tile = blockIdx.x; tile < NTILES; tile += gridDim.x) {
    const int row0 = tile * TILE;

    // stage A: row metadata
    if (tid < TILE) {
      int b = bseg[row0 + tid];
      bseg_s[tid] = b;
      attn_s[tid] = a_arr[row0 + tid] / (anorm[b] + EPS);
    }
    __syncthreads();

    // stage B: u = silu(attn * vrow[b])  ->  LDS f16 (row stride 136)
    {
      int i  = tid >> 2;            // row 0..63
      int f0 = (tid & 3) * 32;      // 32 contiguous features
      float at = attn_s[i];
      const v4f* vr4 = (const v4f*)(vrow + (size_t)bseg_s[i] * FDIM + f0);
      _Float16* ur = u_t + i * USTRIDE + f0;
#pragma unroll
      for (int jj = 0; jj < 4; ++jj) {
        v4f va = vr4[jj * 2 + 0];
        v4f vb = vr4[jj * 2 + 1];
        v8h o;
        o[0] = (_Float16)siluf(at * va[0]); o[1] = (_Float16)siluf(at * va[1]);
        o[2] = (_Float16)siluf(at * va[2]); o[3] = (_Float16)siluf(at * va[3]);
        o[4] = (_Float16)siluf(at * vb[0]); o[5] = (_Float16)siluf(at * vb[1]);
        o[6] = (_Float16)siluf(at * vb[2]); o[7] = (_Float16)siluf(at * vb[3]);
        *(v8h*)(ur + jj * 8) = o;
      }
    }
    __syncthreads();

    // GEMM1: z = silu(u @ W1^T) for columns [n0, n0+16)
#pragma unroll
    for (int mb = 0; mb < 4; ++mb) {
      const int mrow = (mb * 16 + (lane & 15)) * USTRIDE;
      const int koff = (lane >> 4) * 8;   // ISA 16-bit A layout: hi lanes +8 in K
      v8f acc = {};
#pragma unroll
      for (int kc = 0; kc < 4; ++kc) {
        const int base = mrow + kc * 32 + koff;
        v8h lo = *(const v8h*)(u_t + base);        // K = base .. base+7
        v8h hi = *(const v8h*)(u_t + base + 16);   // K = base+16 .. base+23
        v16h af = __builtin_shufflevector(lo, hi, 0,1,2,3,4,5,6,7,
                                                  8,9,10,11,12,13,14,15);
        acc = __builtin_amdgcn_wmma_f32_16x16x32_f16(
            false, af, false, bw1[kc], (short)0, acc, false, false);
      }
      // D layout: vgpr r -> m = r + 8*(lane>=16), n = lane&15
      const int mbase = mb * 16 + (lane >> 4) * 8;
      const int nn = n0 + (lane & 15);
#pragma unroll
      for (int r = 0; r < 8; ++r)
        z_t[(mbase + r) * USTRIDE + nn] = (_Float16)siluf(acc[r]);
    }
    __syncthreads();

    // GEMM2: res = z @ W2^T ; out = h + res
#pragma unroll
    for (int mb = 0; mb < 4; ++mb) {
      const int mrow = (mb * 16 + (lane & 15)) * USTRIDE;
      const int koff = (lane >> 4) * 8;
      v8f acc = {};
#pragma unroll
      for (int kc = 0; kc < 4; ++kc) {
        const int base = mrow + kc * 32 + koff;
        v8h lo = *(const v8h*)(z_t + base);
        v8h hi = *(const v8h*)(z_t + base + 16);
        v16h af = __builtin_shufflevector(lo, hi, 0,1,2,3,4,5,6,7,
                                                  8,9,10,11,12,13,14,15);
        acc = __builtin_amdgcn_wmma_f32_16x16x32_f16(
            false, af, false, bw2[kc], (short)0, acc, false, false);
      }
      const int mbase = mb * 16 + (lane >> 4) * 8;
      const int nn = n0 + (lane & 15);
#pragma unroll
      for (int r = 0; r < 8; ++r) {
        int m = mbase + r;
        float hv = attn_s[m] * vrow[(size_t)bseg_s[m] * FDIM + nn]; // exact h, f32
        // 512MB write-once stream: non-temporal, keep L2 for vrow/weights.
        __builtin_nontemporal_store(hv + acc[r],
                                    &out[(size_t)(row0 + m) * FDIM + nn]);
      }
    }
    __syncthreads();
  }
}

// ---------------------------------------------------------------------------
extern "C" void kernel_launch(void* const* d_in, const int* in_sizes, int n_in,
                              void* d_out, int out_size, void* d_ws, size_t ws_size,
                              hipStream_t stream) {
  const float* x    = (const float*)d_in[0];
  const float* E    = (const float*)d_in[1];
  const int*   bseg = (const int*)d_in[3];     // sorted segment ids
  const float* Wq   = (const float*)d_in[4];
  const float* bq   = (const float*)d_in[5];
  const float* Wk   = (const float*)d_in[6];
  const float* Wv   = (const float*)d_in[7];
  const float* W1   = (const float*)d_in[8];
  const float* W2   = (const float*)d_in[9];
  float* out = (float*)d_out;

  // workspace carve-up (all offsets 256B-aligned)
  char* p = (char*)d_ws;
  float* crow  = (float*)p; p += (size_t)NBATCH * FDIM * 4;  // 2 MB
  float* vrow  = (float*)p; p += (size_t)NBATCH * FDIM * 4;  // 2 MB
  float* a_arr = (float*)p; p += (size_t)NROWS * 4;          // 4 MB
  float* dvec  = (float*)p; p += (size_t)NBATCH * 4;
  float* anorm = (float*)p; p += (size_t)NBATCH * 4;
  _Float16* W1p = (_Float16*)p; p += 8192 * 2;
  _Float16* W2p = (_Float16*)p; p += 8192 * 2;
  float* t0a = (float*)p; p += 128 * 4;
  float* t1a = (float*)p; p += 128 * 4;
  float* u01 = (float*)p; p += 256;

  k_prep0<<<1, 128, 0, stream>>>(Wq, bq, Wk, t0a, t1a, u01);
  k_prep1<<<(NBATCH * FDIM) / 256, 256, 0, stream>>>(E, Wv, t0a, t1a, u01,
                                                     crow, vrow, dvec, anorm);
  k_pack<<<64, 256, 0, stream>>>(W1, W2, W1p, W2p);
  // 7813 waves of 128 rows each, 8 waves/block
  k_dot<<<977, 256, 0, stream>>>(x, bseg, crow, dvec, a_arr, anorm);
  k_mlp<<<2048, 256, 0, stream>>>(a_arr, anorm, bseg, vrow, W1p, W2p, out);
}
